// SSMTrack_52286931862193
// MI455X (gfx1250) — matrix operand order repaired and checked by reading, not compiled, not run on hardware
//
#include <hip/hip_runtime.h>
#include <hip/hip_bf16.h>
#include <math.h>

// Problem sizes (fixed by the reference)
#define BB 8
#define SS 4096
#define HH 1024
#define DD 256
#define MM (BB * SS)   // 32768 flattened rows

typedef __bf16 v16bf __attribute__((ext_vector_type(16)));
typedef float  v8f   __attribute__((ext_vector_type(8)));

union FragBF { v16bf v; uint4 u[2]; };

// tanh(x) = 1 - 2/(exp(2x)+1); v_exp_f32 + v_rcp_f32, saturates to +/-1.
__device__ __forceinline__ float tanh_fast(float x) {
    float e = __builtin_amdgcn_exp2f(x * 2.885390081777927f);  // 2*log2(e)
    float r = __builtin_amdgcn_rcpf(e + 1.0f);
    return __builtin_fmaf(-2.0f, r, 1.0f);
}

// ---------------------------------------------------------------------------
// f32 [rows][cols] -> bf16 [cols][rows] (for W_in, W_out; tiny, one-shot)
// ---------------------------------------------------------------------------
__global__ __launch_bounds__(256) void k_transpose_bf16(const float* __restrict__ in,
                                                        __bf16* __restrict__ out,
                                                        int rows, int cols) {
    int i = blockIdx.x * blockDim.x + threadIdx.x;
    if (i < rows * cols) {
        int r = i / cols, c = i - r * cols;
        out[c * rows + r] = (__bf16)in[i];
    }
}

// ---------------------------------------------------------------------------
// Phase 1 (fused f32->bf16 convert): inp[s*8+b][d] = x[b*S+s][:]@W_in[:,d]+b_in
//   A = x f32 [32768][1024] converted to bf16 fragments in-register
//   B = W_inT bf16 [256][1024] (L2-resident)
// 128 threads (4 waves); block covers 16 rows x full N=256; A reused 4x.
// All loads for a K-chunk issued before the WMMAs -> progressive waits.
// ---------------------------------------------------------------------------
__global__ __launch_bounds__(128) void k_gemm_in(const float* __restrict__ x,
                                                 const __bf16* __restrict__ wT,
                                                 const float* __restrict__ b_in,
                                                 float* __restrict__ inp) {
    const int lane  = threadIdx.x & 31;
    const int wave  = threadIdx.x >> 5;          // 0..3
    const int mt    = blockIdx.x;                // 0..2047
    const int l15   = lane & 15;
    const int hi    = lane >> 4;                 // 0 or 1
    const int mrow  = (mt << 4) + l15;
    const int khA   = hi << 4;                   // A halves {0..15}/{16..31}
    const int khB   = hi << 4;                   // B halves {0..15}/{16..31}

    const float*  arow = x + (size_t)mrow * HH + khA;
    const __bf16* brow = wT + (size_t)(((wave * 4) << 4) + l15) * HH + khB;

    v8f acc[4] = {};
    for (int kk = 0; kk < HH; kk += 32) {
        // Issue every load for this K-chunk first (A f32 + 4 B fragments)
        const float* ap = arow + kk;
        float4 f0 = *(const float4*)(ap);
        float4 f1 = *(const float4*)(ap + 4);
        float4 f2 = *(const float4*)(ap + 8);
        float4 f3 = *(const float4*)(ap + 12);
        FragBF b[4];
#pragma unroll
        for (int j = 0; j < 4; ++j) {
            const __bf16* bp = brow + (size_t)j * (16 * HH) + kk;
            b[j].u[0] = *(const uint4*)(bp);
            b[j].u[1] = *(const uint4*)(bp + 8);
        }
        // Convert A while B loads are in flight
        FragBF a;
        a.v[0]  = (__bf16)f0.x; a.v[1]  = (__bf16)f0.y; a.v[2]  = (__bf16)f0.z; a.v[3]  = (__bf16)f0.w;
        a.v[4]  = (__bf16)f1.x; a.v[5]  = (__bf16)f1.y; a.v[6]  = (__bf16)f1.z; a.v[7]  = (__bf16)f1.w;
        a.v[8]  = (__bf16)f2.x; a.v[9]  = (__bf16)f2.y; a.v[10] = (__bf16)f2.z; a.v[11] = (__bf16)f2.w;
        a.v[12] = (__bf16)f3.x; a.v[13] = (__bf16)f3.y; a.v[14] = (__bf16)f3.z; a.v[15] = (__bf16)f3.w;
#pragma unroll
        for (int j = 0; j < 4; ++j)
            acc[j] = __builtin_amdgcn_wmma_f32_16x16x32_bf16(
                false, a.v, false, b[j].v, (short)0, acc[j], false, false);
    }
    // Epilogue: mbase never crosses a batch boundary -> bidx constant,
    // the 8 store rows differ by exactly 8*DD floats (8 KiB, fits imm offset).
    const int mbase = (mt << 4) + (hi << 3);     // m = b*S + s, multiple of 8
    const int bidx  = mbase >> 12;               // constant over i
    const int row0  = ((mbase & 4095) << 3) + bidx;
#pragma unroll
    for (int j = 0; j < 4; ++j) {
        const int n = ((wave * 4 + j) << 4) + l15;
        const float bias = b_in[n];
        float* p = inp + (size_t)row0 * DD + n;
#pragma unroll
        for (int i = 0; i < 8; ++i)
            p[i * (8 * DD)] = acc[j][i] + bias;
    }
}

// ---------------------------------------------------------------------------
// Phase 2: sequential scan; one workgroup (8 waves) owns the recurrence.
// LDS: W_stateT bf16 128 KiB + state bf16 8 KiB + 2x8 KiB inp double-buffer.
// W fragments live in registers for the whole loop; per step the 8 A-fragments
// are bulk-loaded from LDS, then 16 WMMAs run back-to-back (two interleaved
// accumulator chains). inp_{t+1} prefetched via global_load_async_to_lds_b128
// (ASYNCcnt) while the WMMA chain runs.
// ---------------------------------------------------------------------------
__global__ __launch_bounds__(256) void k_scan(const float* __restrict__ Wst,
                                              const float* __restrict__ b_state,
                                              const float* __restrict__ inp,
                                              __bf16* __restrict__ states) {
    extern __shared__ char smc[];
    __bf16* WT   = (__bf16*)(smc);            // [n][k] 256x256, 131072 B
    __bf16* st   = (__bf16*)(smc + 131072);   // [16][256], 8192 B
    float*  ibuf = (float*)(smc + 139264);    // 2 x 2048 f32

    const int tid  = threadIdx.x;
    const int lane = tid & 31;
    const int wave = tid >> 5;       // 0..7
    const int l15  = lane & 15;
    const int hi   = lane >> 4;
    const int khA  = hi << 4;
    const int khB  = hi << 4;
    const int rbase = hi << 3;

    // Stage W_state (transposed) into LDS as bf16 (coalesced global reads)
    for (int idx = tid; idx < DD * DD; idx += 256) {
        const int k = idx >> 8, n = idx & 255;
        WT[n * DD + k] = (__bf16)Wst[idx];
    }
    for (int idx = tid; idx < 16 * DD; idx += 256) st[idx] = (__bf16)0.0f;

    const int n0 = ((wave * 2) << 4) + l15;
    const int n1 = ((wave * 2 + 1) << 4) + l15;
    const float bias0 = b_state[n0];
    const float bias1 = b_state[n1];

    // Async-load inp for t=0 into buffer 0 (lane -> 16B chunk of LDS)
    {
        const float* g0 = inp + tid * 4;
        const float* g1 = g0 + 1024;
        unsigned l0 = (unsigned)(unsigned long long)(ibuf + tid * 4);
        unsigned l1 = l0 + 4096;
        asm volatile("global_load_async_to_lds_b128 %0, %1, off" :: "v"(l0), "v"(g0) : "memory");
        asm volatile("global_load_async_to_lds_b128 %0, %1, off" :: "v"(l1), "v"(g1) : "memory");
        asm volatile("s_wait_asynccnt 0x0" ::: "memory");
    }
    __syncthreads();

    // Hoist W fragments for this wave's 2 N-tiles into registers (loop-invariant)
    FragBF wb[2][8];
#pragma unroll
    for (int kc = 0; kc < 8; ++kc) {
        const __bf16* bp0 = WT + n0 * DD + kc * 32 + khB;
        wb[0][kc].u[0] = *(const uint4*)(bp0);
        wb[0][kc].u[1] = *(const uint4*)(bp0 + 8);
        const __bf16* bp1 = WT + n1 * DD + kc * 32 + khB;
        wb[1][kc].u[0] = *(const uint4*)(bp1);
        wb[1][kc].u[1] = *(const uint4*)(bp1 + 8);
    }

    const __bf16* stbase = st + l15 * DD + khA;

    for (int t = 0; t < SS; ++t) {
        // Prefetch inp for t+1 into the other LDS buffer (overlaps WMMA chain)
        if (t + 1 < SS) {
            const float* g0 = inp + (size_t)(t + 1) * (8 * DD) + tid * 4;
            const float* g1 = g0 + 1024;
            unsigned l0 = (unsigned)(unsigned long long)(ibuf + ((t + 1) & 1) * 2048 + tid * 4);
            unsigned l1 = l0 + 4096;
            asm volatile("global_load_async_to_lds_b128 %0, %1, off" :: "v"(l0), "v"(g0) : "memory");
            asm volatile("global_load_async_to_lds_b128 %0, %1, off" :: "v"(l1), "v"(g1) : "memory");
        }

        // Pull this step's inp rows into registers early (overlaps WMMA chain)
        float i0[8] = {0,0,0,0,0,0,0,0}, i1[8] = {0,0,0,0,0,0,0,0};
        if (lane < 16) {
            const float* it = ibuf + (t & 1) * 2048;
#pragma unroll
            for (int i = 0; i < 8; ++i) {
                i0[i] = it[i * DD + n0];
                i1[i] = it[i * DD + n1];
            }
        }

        // Bulk-load all 8 A-fragments, then run the WMMA chain uninterrupted
        FragBF af[8];
#pragma unroll
        for (int kc = 0; kc < 8; ++kc) {
            const __bf16* ap = stbase + kc * 32;
            af[kc].u[0] = *(const uint4*)(ap);
            af[kc].u[1] = *(const uint4*)(ap + 8);
        }
        v8f acc0 = {}, acc1 = {};
#pragma unroll
        for (int kc = 0; kc < 8; ++kc) {
            acc0 = __builtin_amdgcn_wmma_f32_16x16x32_bf16(
                false, af[kc].v, false, wb[0][kc].v, (short)0, acc0, false, false);
            acc1 = __builtin_amdgcn_wmma_f32_16x16x32_bf16(
                false, af[kc].v, false, wb[1][kc].v, (short)0, acc1, false, false);
        }

        // state = tanh(inp_t + state@W + b); pad rows (lanes>=16) have inp=0
        float v0[8], v1[8];
#pragma unroll
        for (int i = 0; i < 8; ++i) {
            v0[i] = tanh_fast(acc0[i] + bias0 + i0[i]);
            v1[i] = tanh_fast(acc1[i] + bias1 + i1[i]);
        }
        __syncthreads();   // all waves done reading old state + inp buffer
#pragma unroll
        for (int i = 0; i < 8; ++i) {
            const int r = rbase + i;
            st[r * DD + n0] = (__bf16)v0[i];
            st[r * DD + n1] = (__bf16)v1[i];
        }
        if (lane < 16) {   // persist real rows for phase 3 (bf16)
            __bf16* so = states + (size_t)t * (8 * DD);
#pragma unroll
            for (int i = 0; i < 8; ++i) {
                so[i * DD + n0] = (__bf16)v0[i];
                so[i * DD + n1] = (__bf16)v1[i];
            }
        }
        asm volatile("s_wait_asynccnt 0x0" ::: "memory");  // t+1 buffer landed
        __syncthreads();   // new state + prefetched inp visible to all waves
    }
}

// ---------------------------------------------------------------------------
// Phase 3: out[b][s][h] = states[s*8+b][:] @ W_out[:,h] + b_out[h]
//   A = states bf16 [32768][256] (L2-resident across the 4 N-groups)
//   B = W_outT bf16 [1024][256]
// All loads per K-chunk issued before WMMAs; epilogue uses constant sidx and
// batch-stride pointer bumps (bidx == i) instead of per-row 64-bit remap math.
// ---------------------------------------------------------------------------
__global__ __launch_bounds__(128) void k_gemm_out(const __bf16* __restrict__ stg,
                                                  const __bf16* __restrict__ woT,
                                                  const float* __restrict__ b_out,
                                                  float* __restrict__ out) {
    const int lane = threadIdx.x & 31;
    const int wave = threadIdx.x >> 5;
    const int mt   = blockIdx.x;
    const int ng   = blockIdx.y;
    const int l15  = lane & 15;
    const int hi   = lane >> 4;
    const int mrow = (mt << 4) + l15;
    const int khA  = hi << 4;
    const int khB  = hi << 4;

    const __bf16* arow = stg + (size_t)mrow * DD + khA;
    const __bf16* brow = woT + (size_t)((ng << 8) + ((wave * 4) << 4) + l15) * DD + khB;

    v8f acc[4] = {};
#pragma unroll
    for (int kk = 0; kk < DD; kk += 32) {
        FragBF a, b[4];
        const __bf16* ap = arow + kk;
        a.u[0] = *(const uint4*)(ap);
        a.u[1] = *(const uint4*)(ap + 8);
#pragma unroll
        for (int j = 0; j < 4; ++j) {
            const __bf16* bp = brow + (size_t)j * (16 * DD) + kk;
            b[j].u[0] = *(const uint4*)(bp);
            b[j].u[1] = *(const uint4*)(bp + 8);
        }
#pragma unroll
        for (int j = 0; j < 4; ++j)
            acc[j] = __builtin_amdgcn_wmma_f32_16x16x32_bf16(
                false, a.v, false, b[j].v, (short)0, acc[j], false, false);
    }
    // m = mbase+i = s*8 + b with mbase a multiple of 8 -> sidx constant, b = i
    const int mbase = (mt << 4) + (hi << 3);
    const int sidx  = mbase >> 3;
#pragma unroll
    for (int j = 0; j < 4; ++j) {
        const int n = (ng << 8) + ((wave * 4 + j) << 4) + l15;
        const float bias = b_out[n];
        float* p = out + (size_t)sidx * HH + n;
#pragma unroll
        for (int i = 0; i < 8; ++i) {
            *p = acc[j][i] + bias;
            p += (size_t)SS * HH;   // next batch, same (s, h)
        }
    }
}

// ---------------------------------------------------------------------------
// Host launcher
// ---------------------------------------------------------------------------
extern "C" void kernel_launch(void* const* d_in, const int* in_sizes, int n_in,
                              void* d_out, int out_size, void* d_ws, size_t ws_size,
                              hipStream_t stream) {
    const float* x       = (const float*)d_in[0];
    const float* W_in    = (const float*)d_in[1];
    const float* b_in    = (const float*)d_in[2];
    const float* W_state = (const float*)d_in[3];
    const float* b_state = (const float*)d_in[4];
    const float* W_out   = (const float*)d_in[5];
    const float* b_out   = (const float*)d_in[6];
    float* out = (float*)d_out;

    // Workspace layout (bytes):
    //   W_inT   : 256*1024*2   = 512 KiB   @ 0
    //   W_outT  : 1024*256*2   = 512 KiB   @ 524288
    //   inp f32 : 32768*256*4  = 32 MiB    @ 1048576
    //   states  : 32768*256*2  = 16 MiB    @ 34603008
    char* ws = (char*)d_ws;
    __bf16* winT   = (__bf16*)(ws);
    __bf16* woutT  = (__bf16*)(ws + (size_t)524288);
    float*  inp    = (float*) (ws + (size_t)1048576);
    __bf16* states = (__bf16*)(ws + (size_t)34603008);

    k_transpose_bf16<<<(HH * DD + 255) / 256, 256, 0, stream>>>(W_in, winT, HH, DD);
    k_transpose_bf16<<<(DD * HH + 255) / 256, 256, 0, stream>>>(W_out, woutT, DD, HH);

    k_gemm_in<<<MM / 16, 128, 0, stream>>>(x, winT, b_in, inp);

    const size_t smem = 139264 + 2 * 2048 * sizeof(float);  // 152 KiB
    k_scan<<<1, 256, smem, stream>>>(W_state, b_state, inp, states);

    k_gemm_out<<<dim3(MM / 16, 4), 128, 0, stream>>>(states, woutT, b_out, out);
}
